// Experts_45011257262413
// MI455X (gfx1250) — matrix operand order repaired and checked by reading, not compiled, not run on hardware
//
#include <hip/hip_runtime.h>
#include <hip/hip_bf16.h>
#include <math.h>

// Problem sizes (fixed by the reference).
#define E_ 16
#define N_ 2048
#define D_ 1024
#define H_ 4096

// Tiling: 128x128 workgroup tile, K-chunk 32 (one bf16 WMMA K step).
#define TM 128
#define TN 128
#define KC 32
// Padded LDS row stride (bf16 elements): 40*2=80B keeps b128 fragment loads
// 16B-aligned while spreading rows across 16 distinct banks (20*row mod 64).
#define LDK 40

typedef __bf16 bf16_t;
typedef __attribute__((ext_vector_type(16))) __bf16 v16bf;
typedef __attribute__((ext_vector_type(8)))  __bf16 v8bf;
typedef __attribute__((ext_vector_type(4)))  __bf16 v4bf;
typedef __attribute__((ext_vector_type(8)))  float   v8f;

__device__ __forceinline__ float gelu_erf(float x) {
    // torch nn.GELU default: 0.5*x*(1+erf(x/sqrt(2)))
    return 0.5f * x * (1.0f + erff(x * 0.70710678118654752440f));
}

// A fragment (16x32 bf16) from LDS tile stored row-major [m][LDK].
// ISA layout: lane L in 0..15 holds row M=L, K = {kh*8..kh*8+7} u {16+kh*8..+7},
// kh = lane>>4; v16bf element order matches VGPR packing (2 bf16 per VGPR).
__device__ __forceinline__ v16bf load_a_frag(const bf16_t* lds, int m_base, int lane) {
    const int row = m_base + (lane & 15);
    const int kh  = (lane >> 4) & 1;
    const v8bf lo = *(const v8bf*)(lds + row * LDK + kh * 8);
    const v8bf hi = *(const v8bf*)(lds + row * LDK + 16 + kh * 8);
    v16bf a;
#pragma unroll
    for (int i = 0; i < 8; ++i) { a[i] = lo[i]; a[i + 8] = hi[i]; }
    return a;
}

// B fragment (32x16 bf16) from LDS tile stored transposed [n][LDK].
// ISA layout: lane L in 0..15 holds column N=L, K = {kh*8..kh*8+7} u {16+kh*8..+7}.
__device__ __forceinline__ v16bf load_b_frag(const bf16_t* lds, int n_base, int lane) {
    const int col = n_base + (lane & 15);
    const int kh  = (lane >> 4) & 1;
    const v8bf lo = *(const v8bf*)(lds + col * LDK + kh * 8);
    const v8bf hi = *(const v8bf*)(lds + col * LDK + 16 + kh * 8);
    v16bf b;
#pragma unroll
    for (int i = 0; i < 8; ++i) { b[i] = lo[i]; b[i + 8] = hi[i]; }
    return b;
}

// ---- global->register prefetch + register->LDS commit helpers --------------

// A tile (f32 source): 128x32 floats = 1024 float4, 4 per thread.
__device__ __forceinline__ void fetch_a_f32(const float* __restrict__ src, int row0,
                                            int ld, int k0, int t, float4 v[4]) {
#pragma unroll
    for (int i = 0; i < 4; ++i) {
        const int idx = t + i * 256;
        const int r   = idx >> 3;            // 8 float4 per row
        const int c   = (idx & 7) << 2;
        v[i] = *(const float4*)(src + (size_t)(row0 + r) * ld + k0 + c);
    }
}
__device__ __forceinline__ void commit_a_f32(bf16_t* lds, int t, const float4 v[4]) {
#pragma unroll
    for (int i = 0; i < 4; ++i) {
        const int idx = t + i * 256;
        const int r   = idx >> 3;
        const int c   = (idx & 7) << 2;
        v4bf p;
        p[0] = (bf16_t)v[i].x; p[1] = (bf16_t)v[i].y;
        p[2] = (bf16_t)v[i].z; p[3] = (bf16_t)v[i].w;
        *(v4bf*)(lds + r * LDK + c) = p;     // single ds_store_b64
    }
}

// B tile (f32 source, transposed into LDS): 32x128 floats = 1024 float4.
__device__ __forceinline__ void fetch_b_f32(const float* __restrict__ src, int col0,
                                            int ld, int k0, int t, float4 v[4]) {
#pragma unroll
    for (int i = 0; i < 4; ++i) {
        const int idx = t + i * 256;
        const int kr  = idx >> 5;            // 32 float4 per k-row
        const int cq  = (idx & 31) << 2;
        v[i] = *(const float4*)(src + (size_t)(k0 + kr) * ld + col0 + cq);
    }
}
__device__ __forceinline__ void commit_b_f32(bf16_t* lds, int t, const float4 v[4]) {
#pragma unroll
    for (int i = 0; i < 4; ++i) {
        const int idx = t + i * 256;
        const int kr  = idx >> 5;
        const int cq  = (idx & 31) << 2;
        lds[(cq + 0) * LDK + kr] = (bf16_t)v[i].x;
        lds[(cq + 1) * LDK + kr] = (bf16_t)v[i].y;
        lds[(cq + 2) * LDK + kr] = (bf16_t)v[i].z;
        lds[(cq + 3) * LDK + kr] = (bf16_t)v[i].w;
    }
}

// A tile (bf16 source, straight copy): 128x32 bf16 = 512 8-elt chunks, 2/thread.
__device__ __forceinline__ void fetch_a_bf16(const bf16_t* __restrict__ src, int row0,
                                             int ld, int k0, int t, v8bf v[2]) {
#pragma unroll
    for (int i = 0; i < 2; ++i) {
        const int idx = t + i * 256;
        const int r   = idx >> 2;            // 4 chunks per row
        const int c   = (idx & 3) << 3;
        v[i] = *(const v8bf*)(src + (size_t)(row0 + r) * ld + k0 + c);
    }
}
__device__ __forceinline__ void commit_a_bf16(bf16_t* lds, int t, const v8bf v[2]) {
#pragma unroll
    for (int i = 0; i < 2; ++i) {
        const int idx = t + i * 256;
        const int r   = idx >> 2;
        const int c   = (idx & 3) << 3;
        *(v8bf*)(lds + r * LDK + c) = v[i];  // ds_store_b128
    }
}

// ---------------------------------------------------------------------------
// Kernel 1: hidden[e] = gelu( X[e](2048x1024) * W1[e](1024x4096) ), bf16 out.
// Double-buffered LDS, global prefetch overlapped with WMMA.
// ---------------------------------------------------------------------------
__global__ __launch_bounds__(256, 2)
void mlp_gemm1_gelu(const float* __restrict__ X, const float* __restrict__ W1,
                    bf16_t* __restrict__ Hid) {
    __shared__ bf16_t lA[2][TM * LDK];   // [m][k]
    __shared__ bf16_t lB[2][TN * LDK];   // [h][k] (transposed)

    const int e  = blockIdx.z;
    const int n0 = blockIdx.y * TM;
    const int h0 = blockIdx.x * TN;
    const float* Xe  = X  + (size_t)e * N_ * D_;
    const float* W1e = W1 + (size_t)e * D_ * H_;
    bf16_t*      He  = Hid + (size_t)e * N_ * H_;

    const int t    = threadIdx.x;
    const int lane = t & 31;
    const int wave = t >> 5;
    const int wm   = wave >> 1;   // 0..3 -> rows wm*32
    const int wn   = wave & 1;    // 0..1 -> cols wn*64

    v8f acc[2][4] = {};

    float4 va[4], vb[4];
    fetch_a_f32(Xe, n0, D_, 0, t, va);
    fetch_b_f32(W1e, h0, H_, 0, t, vb);
    commit_a_f32(lA[0], t, va);
    commit_b_f32(lB[0], t, vb);
    __syncthreads();

    const int NK = D_ / KC;
    for (int kc = 0; kc < NK; ++kc) {
        const int cur = kc & 1;
        const bool more = (kc + 1) < NK;
        if (more) {                       // issue next chunk's global loads early
            fetch_a_f32(Xe, n0, D_, (kc + 1) * KC, t, va);
            fetch_b_f32(W1e, h0, H_, (kc + 1) * KC, t, vb);
        }

        v16bf af[2], bfr[4];
#pragma unroll
        for (int i = 0; i < 2; ++i) af[i] = load_a_frag(lA[cur], wm * 32 + i * 16, lane);
#pragma unroll
        for (int j = 0; j < 4; ++j) bfr[j] = load_b_frag(lB[cur], wn * 64 + j * 16, lane);
#pragma unroll
        for (int i = 0; i < 2; ++i)
#pragma unroll
            for (int j = 0; j < 4; ++j)
                acc[i][j] = __builtin_amdgcn_wmma_f32_16x16x32_bf16(
                    false, af[i], false, bfr[j], (short)0, acc[i][j], false, false);

        if (more) {                       // commit prefetched chunk to other buffer
            commit_a_f32(lA[cur ^ 1], t, va);
            commit_b_f32(lB[cur ^ 1], t, vb);
        }
        __syncthreads();
    }

    // Epilogue: GELU, store bf16 hidden.
    // C/D layout: lane&15 = column, row = r + 8*(lane>>4) within each 16x16 tile.
    const int cn = lane & 15;
    const int rh = (lane >> 4) * 8;
#pragma unroll
    for (int i = 0; i < 2; ++i) {
#pragma unroll
        for (int j = 0; j < 4; ++j) {
            const int rbase = n0 + wm * 32 + i * 16 + rh;
            const int col   = h0 + wn * 64 + j * 16 + cn;
#pragma unroll
            for (int r = 0; r < 8; ++r) {
                const float g = gelu_erf(acc[i][j][r]);
                He[(size_t)(rbase + r) * H_ + col] = (bf16_t)g;
            }
        }
    }
}

// ---------------------------------------------------------------------------
// Kernel 2: out[e] = hidden[e](2048x4096 bf16) * W2[e](4096x1024), f32 out.
// ---------------------------------------------------------------------------
__global__ __launch_bounds__(256, 2)
void mlp_gemm2(const bf16_t* __restrict__ Hid, const float* __restrict__ W2,
               float* __restrict__ Out) {
    __shared__ bf16_t lA[2][TM * LDK];   // [n][k]
    __shared__ bf16_t lB[2][TN * LDK];   // [d][k] (transposed)

    const int e  = blockIdx.z;
    const int n0 = blockIdx.y * TM;
    const int d0 = blockIdx.x * TN;
    const bf16_t* He  = Hid + (size_t)e * N_ * H_;
    const float*  W2e = W2 + (size_t)e * H_ * D_;
    float*        Oe  = Out + (size_t)e * N_ * D_;

    const int t    = threadIdx.x;
    const int lane = t & 31;
    const int wave = t >> 5;
    const int wm   = wave >> 1;
    const int wn   = wave & 1;

    v8f acc[2][4] = {};

    v8bf va[2];
    float4 vb[4];
    fetch_a_bf16(He, n0, H_, 0, t, va);
    fetch_b_f32(W2e, d0, D_, 0, t, vb);
    commit_a_bf16(lA[0], t, va);
    commit_b_f32(lB[0], t, vb);
    __syncthreads();

    const int NK = H_ / KC;
    for (int kc = 0; kc < NK; ++kc) {
        const int cur = kc & 1;
        const bool more = (kc + 1) < NK;
        if (more) {
            fetch_a_bf16(He, n0, H_, (kc + 1) * KC, t, va);
            fetch_b_f32(W2e, d0, D_, (kc + 1) * KC, t, vb);
        }

        v16bf af[2], bfr[4];
#pragma unroll
        for (int i = 0; i < 2; ++i) af[i] = load_a_frag(lA[cur], wm * 32 + i * 16, lane);
#pragma unroll
        for (int j = 0; j < 4; ++j) bfr[j] = load_b_frag(lB[cur], wn * 64 + j * 16, lane);
#pragma unroll
        for (int i = 0; i < 2; ++i)
#pragma unroll
            for (int j = 0; j < 4; ++j)
                acc[i][j] = __builtin_amdgcn_wmma_f32_16x16x32_bf16(
                    false, af[i], false, bfr[j], (short)0, acc[i][j], false, false);

        if (more) {
            commit_a_bf16(lA[cur ^ 1], t, va);
            commit_b_f32(lB[cur ^ 1], t, vb);
        }
        __syncthreads();
    }

    // Epilogue: store f32 output.
    const int cn = lane & 15;
    const int rh = (lane >> 4) * 8;
#pragma unroll
    for (int i = 0; i < 2; ++i) {
#pragma unroll
        for (int j = 0; j < 4; ++j) {
            const int rbase = n0 + wm * 32 + i * 16 + rh;
            const int col   = d0 + wn * 64 + j * 16 + cn;
#pragma unroll
            for (int r = 0; r < 8; ++r)
                Oe[(size_t)(rbase + r) * D_ + col] = acc[i][j][r];
        }
    }
}

extern "C" void kernel_launch(void* const* d_in, const int* in_sizes, int n_in,
                              void* d_out, int out_size, void* d_ws, size_t ws_size,
                              hipStream_t stream) {
    const float* x  = (const float*)d_in[0];   // (E,N,D) f32
    const float* w1 = (const float*)d_in[1];   // (E,D,H) f32
    const float* w2 = (const float*)d_in[2];   // (E,H,D) f32
    float* out = (float*)d_out;                // (E,N,D) f32
    bf16_t* hid = (bf16_t*)d_ws;               // (E,N,H) bf16 scratch: 268 MB

    dim3 blk(256);
    mlp_gemm1_gelu<<<dim3(H_ / TN, N_ / TM, E_), blk, 0, stream>>>(x, w1, hid);
    mlp_gemm2   <<<dim3(D_ / TN, N_ / TM, E_), blk, 0, stream>>>(hid, w2, out);
}
